// RNNController_70583492543029
// MI455X (gfx1250) — matrix operand order repaired and checked by reading, not compiled, not run on hardware
//
#include <hip/hip_runtime.h>
#include <math.h>

// ---------------------------------------------------------------------------
// RNN view-pointer controller, fully fused persistent kernel for gfx1250.
//   - algebraic fusion of the point backbone
//   - 3 GEMMs/step via v_wmma_f32_16x16x32_f16, register-blocked tiles
//   - batch_data gather via Tensor Data Mover (tensor_load_to_lds + TENSORcnt)
//   - one persistent kernel, grid barrier via agent atomics + s_sleep
// ---------------------------------------------------------------------------

#define BS    512
#define VL    64
#define PTS   256
#define TMAX  8
#define NBLK  64
#define NTHR  256
#define NWAVE 512            // NBLK*NTHR/32
#define NEGV  -1e9f

typedef __attribute__((ext_vector_type(16))) _Float16 v16h;
typedef __attribute__((ext_vector_type(8)))  float    v8f;
typedef __attribute__((ext_vector_type(4)))  unsigned v4u;
typedef __attribute__((ext_vector_type(8)))  int      v8i;
typedef __attribute__((ext_vector_type(4)))  int      v4i;

union Frag { unsigned u[8]; v16h h; };

// ----------------------------- workspace layout ----------------------------
__host__ __device__ constexpr size_t alignUp(size_t x) { return (x + 255) & ~size_t(255); }

constexpr size_t O_BAR  = 0;                                  // 2 uints (count, gen)
constexpr size_t O_SUM  = 256;                                // 1 float  (sum of w1_w)
constexpr size_t O_V    = 512;                                // BS int32 (current view)
constexpr size_t O_MASK = alignUp(O_V    + BS * 4);           // BS*VL u8
constexpr size_t O_B1   = alignUp(O_MASK + BS * VL);          // blend1 f32 [VL][256]
constexpr size_t O_ENC  = alignUp(O_B1   + VL * 256 * 4);     // enc_emb f16 [VL][128]
constexpr size_t O_W2T  = alignUp(O_ENC  + VL * 128 * 2);     // w2_w^T  f16 [512][1024]
constexpr size_t O_WIH  = alignUp(O_W2T  + 512 * 1024 * 2);   // W_ih    f16 [2048][640]
constexpr size_t O_WHH  = alignUp(O_WIH  + 2048 * 640 * 2);   // W_hh    f16 [2048][512]
constexpr size_t O_WAT  = alignUp(O_WHH  + 2048 * 512 * 2);   // W2_w^T  f16 [256][512]
constexpr size_t O_ACT  = alignUp(O_WAT  + 256 * 512 * 2);    // relu(feat) f16 [BS][1024]
constexpr size_t O_DEC  = alignUp(O_ACT  + BS * 1024 * 2);    // dec_in  f16 [BS][640]
constexpr size_t O_H    = alignUp(O_DEC  + BS * 640 * 2);     // h       f16 [BS][512]
constexpr size_t O_C    = alignUp(O_H    + BS * 512 * 2);     // c       f32 [BS][512]
constexpr size_t O_G    = alignUp(O_C    + BS * 512 * 4);     // gates   f32 [BS][2048]
constexpr size_t O_BL2  = alignUp(O_G    + BS * 2048 * 4);    // blend2  f32 [BS][256]
constexpr size_t WS_END = O_BL2 + BS * 256 * 4;

// ----------------------------- helpers -------------------------------------
__device__ __forceinline__ float sigmoidf_(float x) { return 1.0f / (1.0f + __expf(-x)); }

// A-matrix fragment (16x32 f16), A row-major [M][lda].  (ISA 7.12.2)
__device__ __forceinline__ v16h load_a16(const _Float16* A, int lda, int row0, int k0, int lane) {
    int m    = row0 + (lane & 15);
    int kmid = (lane >> 4) & 1;
    const unsigned* base = (const unsigned*)(A + (size_t)m * lda + k0);
    Frag f;
#pragma unroll
    for (int r = 0; r < 8; ++r) {
        int pair = (kmid << 3) + r;   // (k/2) offset
        f.u[r] = base[pair];
    }
    return f.h;
}

// B-matrix fragment (32x16 f16); BT row-major [N][ldb], (k,n) at BT[n*ldb+k].
__device__ __forceinline__ v16h load_b16(const _Float16* BT, int ldb, int n0, int k0, int lane) {
    int n  = n0 + (lane & 15);
    int kg = (lane >> 4) & 1;
    const unsigned* base = (const unsigned*)(BT + (size_t)n * ldb + k0 + kg * 16);
    Frag f;
#pragma unroll
    for (int r = 0; r < 8; ++r) f.u[r] = base[r];
    return f.h;
}

__device__ __forceinline__ v8f wmma16(v16h a, v16h b, v8f c) {
    return __builtin_amdgcn_wmma_f32_16x16x32_f16(false, a, false, b, (short)0, c, false, false);
}

// grid-wide barrier: agent-scope atomics, s_sleep while spinning
__device__ __forceinline__ void grid_barrier(unsigned* bar) {
    __syncthreads();
    if (threadIdx.x == 0) {
        __threadfence();
        unsigned gen = __hip_atomic_load(&bar[1], __ATOMIC_ACQUIRE, __HIP_MEMORY_SCOPE_AGENT);
        unsigned arrived =
            __hip_atomic_fetch_add(&bar[0], 1u, __ATOMIC_ACQ_REL, __HIP_MEMORY_SCOPE_AGENT);
        if (arrived == (unsigned)(NBLK - 1)) {
            __hip_atomic_store(&bar[0], 0u, __ATOMIC_RELAXED, __HIP_MEMORY_SCOPE_AGENT);
            __hip_atomic_fetch_add(&bar[1], 1u, __ATOMIC_ACQ_REL, __HIP_MEMORY_SCOPE_AGENT);
        } else {
            while (__hip_atomic_load(&bar[1], __ATOMIC_ACQUIRE, __HIP_MEMORY_SCOPE_AGENT) == gen)
                __builtin_amdgcn_s_sleep(2);
        }
        __threadfence();
    }
    __syncthreads();
}

// ----------------------------- setup kernel --------------------------------
__global__ void rnnctl_setup(const float* __restrict__ enc,     // [64][128]
                             const float* __restrict__ w1w,     // [256]
                             const float* __restrict__ w2w,     // [1024][512]
                             const float* __restrict__ wih,     // [2048][640]
                             const float* __restrict__ whh,     // [2048][512]
                             const float* __restrict__ W1w,     // [128][256]
                             const float* __restrict__ W2w,     // [512][256]
                             char* __restrict__ ws) {
    size_t tid    = (size_t)blockIdx.x * blockDim.x + threadIdx.x;
    size_t stride = (size_t)gridDim.x * blockDim.x;

    if (tid == 0) {
        float s = 0.f;
        for (int i = 0; i < PTS; ++i) s += w1w[i];
        *(float*)(ws + O_SUM) = s;
    }

    _Float16* w2T   = (_Float16*)(ws + O_W2T);
    _Float16* wih16 = (_Float16*)(ws + O_WIH);
    _Float16* whh16 = (_Float16*)(ws + O_WHH);
    _Float16* watT  = (_Float16*)(ws + O_WAT);
    _Float16* enc16 = (_Float16*)(ws + O_ENC);
    float*    b1    = (float*)(ws + O_B1);

    for (size_t i = tid; i < (size_t)1024 * 512; i += stride) {          // w2 transpose
        int k = (int)(i >> 9), n = (int)(i & 511);
        w2T[(size_t)n * 1024 + k] = (_Float16)w2w[i];
    }
    for (size_t i = tid; i < (size_t)2048 * 640; i += stride) wih16[i] = (_Float16)wih[i];
    for (size_t i = tid; i < (size_t)2048 * 512; i += stride) whh16[i] = (_Float16)whh[i];
    for (size_t i = tid; i < (size_t)512 * 256; i += stride) {           // W2 transpose
        int k = (int)(i >> 8), n = (int)(i & 255);
        watT[(size_t)n * 512 + k] = (_Float16)W2w[i];
    }
    for (size_t i = tid; i < (size_t)VL * 128; i += stride) enc16[i] = (_Float16)enc[i];
    for (size_t i = tid; i < (size_t)VL * 256; i += stride) {            // blend1
        int u = (int)(i >> 8), j = (int)(i & 255);
        float s = 0.f;
        for (int k = 0; k < 128; ++k) s += enc[u * 128 + k] * W1w[k * 256 + j];
        b1[i] = s;
    }
}

// ----------------------------- persistent kernel ---------------------------
__global__ void __launch_bounds__(NTHR)
rnnctl_persistent(const float* __restrict__ bd,      // batch_data [512][64][256][3]
                  const float* __restrict__ W0w,     // [3][1024]
                  const float* __restrict__ W0b,     // [1024]
                  const float* __restrict__ w1w,     // [256]
                  const float* __restrict__ w1b,     // [1]
                  const float* __restrict__ w2b,     // [512]
                  const float* __restrict__ bih,     // [2048]
                  const float* __restrict__ bhh,     // [2048]
                  const float* __restrict__ vt,      // [256]
                  char* __restrict__ ws,
                  float* __restrict__ out) {         // [8*512 views | 8*512*64 logp]
    __shared__ __align__(16) float ptile[NTHR / 32][PTS * 3];  // 3KB per wave (TDM dest)

    unsigned*  bar   = (unsigned*)(ws + O_BAR);
    int*       vptr  = (int*)(ws + O_V);
    unsigned char* mask = (unsigned char*)(ws + O_MASK);
    const float*   b1    = (const float*)(ws + O_B1);
    const _Float16* enc16 = (const _Float16*)(ws + O_ENC);
    const _Float16* w2T   = (const _Float16*)(ws + O_W2T);
    const _Float16* wih16 = (const _Float16*)(ws + O_WIH);
    const _Float16* whh16 = (const _Float16*)(ws + O_WHH);
    const _Float16* watT  = (const _Float16*)(ws + O_WAT);
    _Float16* actA  = (_Float16*)(ws + O_ACT);
    _Float16* dec   = (_Float16*)(ws + O_DEC);
    _Float16* hbuf  = (_Float16*)(ws + O_H);
    float*    cbuf  = (float*)(ws + O_C);
    float*    gates = (float*)(ws + O_G);
    float*    bl2   = (float*)(ws + O_BL2);

    const int tid   = blockIdx.x * NTHR + threadIdx.x;
    const int wave  = tid >> 5;
    const int wslot = (threadIdx.x >> 5);
    const int lane  = threadIdx.x & 31;
    const float sumw1 = *(const float*)(ws + O_SUM);
    const float wb1   = w1b[0];

    float* out_views = out;
    float* out_logp  = out + TMAX * BS;

    const v8f vzero = {};

    for (int t = 0; t < TMAX; ++t) {
        // ---- Stage A: TDM gather of batch_data[b, v[b]] -> LDS, weighted
        //      point reduction, feat = s@W0 + bias, relu -> actA; enc slot.
        {
            const int b  = wave;                       // one wave per batch row
            const int vv = vptr[b];
            if (lane == 0) mask[b * VL + vv] = 1;

            // --- build 1-D tensor descriptor (D#) and fire the TDM ---
            unsigned ldsoff = (unsigned)(size_t)(void*)&ptile[wslot][0];
            unsigned long long ga =
                (unsigned long long)(size_t)(const void*)(bd + (((size_t)b * VL + vv) * PTS) * 3);
            v4u g0;
            g0[0] = 1u;                                   // count=1, user desc
            g0[1] = ldsoff;                               // lds_addr
            g0[2] = (unsigned)ga;                         // global_addr[31:0]
            g0[3] = (unsigned)(ga >> 32) | 0x80000000u;   // global_addr[56:32] | type=2
            v8i g1;
            g1[0] = (int)(2u << 16);                      // data_size=4B; no multicast
            g1[1] = (int)((PTS * 3u) << 16);              // tensor_dim0 = 768 (lo16)
            g1[2] = (int)(1u << 16);                      // tensor_dim1 = 1
            g1[3] = (int)((PTS * 3u) << 16);              // tile_dim0 = 768
            g1[4] = 0;                                    // tile_dim1/2 = 0 (1-D)
            g1[5] = (int)(PTS * 3u);                      // tensor_dim0_stride = 768
            g1[6] = 0;
            g1[7] = 0;
            v4i z4 = {0, 0, 0, 0};
            v8i z8 = {0, 0, 0, 0, 0, 0, 0, 0};
            __builtin_amdgcn_tensor_load_to_lds(g0, g1, z4, z4, z8, 0);
            __builtin_amdgcn_s_wait_tensorcnt(0);

            const float* vd = &ptile[wslot][0];
            float a0 = 0.f, a1 = 0.f, a2 = 0.f;
            for (int p = lane; p < PTS; p += 32) {
                float w = w1w[p];
                a0 += w * vd[p * 3 + 0];
                a1 += w * vd[p * 3 + 1];
                a2 += w * vd[p * 3 + 2];
            }
#pragma unroll
            for (int off = 16; off > 0; off >>= 1) {
                a0 += __shfl_xor(a0, off);
                a1 += __shfl_xor(a1, off);
                a2 += __shfl_xor(a2, off);
            }
            for (int f = lane; f < 1024; f += 32) {
                float ft = a0 * W0w[f] + a1 * W0w[1024 + f] + a2 * W0w[2048 + f]
                         + sumw1 * W0b[f] + wb1;
                actA[(size_t)b * 1024 + f] = (_Float16)(ft > 0.f ? ft : 0.f);
            }
            const _Float16* ev = enc16 + (size_t)vv * 128;
            for (int j = lane; j < 128; j += 32) dec[(size_t)b * 640 + 512 + j] = ev[j];
        }
        grid_barrier(bar);

        // ---- Stage B: view_feat = relu(actA @ w2 + w2_b) -> dec_in[:,0:512]
        //      1x2 register blocking: one A frag feeds two wmma per k-step.
        {
            int tm  = wave >> 4;                       // 0..31
            int tn0 = (wave & 15) * 2;                 // 0..30
            v8f acc0 = vzero, acc1 = vzero;
            for (int k0 = 0; k0 < 1024; k0 += 32) {
                v16h a  = load_a16(actA, 1024, tm * 16, k0, lane);
                v16h b0 = load_b16(w2T, 1024, tn0 * 16, k0, lane);
                v16h b1f = load_b16(w2T, 1024, (tn0 + 1) * 16, k0, lane);
                acc0 = wmma16(a, b0, acc0);
                acc1 = wmma16(a, b1f, acc1);
            }
            int mbase = tm * 16 + ((lane >> 4) & 1) * 8;
#pragma unroll
            for (int j = 0; j < 2; ++j) {
                int n = (tn0 + j) * 16 + (lane & 15);
                float bias = w2b[n];
                v8f acc = j ? acc1 : acc0;
#pragma unroll
                for (int r = 0; r < 8; ++r) {
                    float v = acc[r] + bias;
                    dec[(size_t)(mbase + r) * 640 + n] = (_Float16)(v > 0.f ? v : 0.f);
                }
            }
        }
        grid_barrier(bar);

        // ---- Stage C: gates = dec_in @ W_ih^T + h @ W_hh^T + b_ih + b_hh
        //      2x4 register blocking: 32x64 output per wave, 8 wmma per k-step
        //      off 2 A-frag + 4 B-frag loads.
        {
            int gm  = wave >> 5;                       // 0..15 -> tm pair
            int gn  = wave & 31;                       // 0..31 -> 4 tn tiles
            int tm0 = gm * 2, tn0 = gn * 4;
            v8f acc[8];
#pragma unroll
            for (int i = 0; i < 8; ++i) acc[i] = vzero;

            for (int k0 = 0; k0 < 640; k0 += 32) {
                v16h a0 = load_a16(dec, 640, tm0 * 16, k0, lane);
                v16h a1 = load_a16(dec, 640, (tm0 + 1) * 16, k0, lane);
#pragma unroll
                for (int j = 0; j < 4; ++j) {
                    v16h b = load_b16(wih16, 640, (tn0 + j) * 16, k0, lane);
                    acc[j]     = wmma16(a0, b, acc[j]);
                    acc[4 + j] = wmma16(a1, b, acc[4 + j]);
                }
            }
            for (int k0 = 0; k0 < 512; k0 += 32) {
                v16h a0 = load_a16(hbuf, 512, tm0 * 16, k0, lane);
                v16h a1 = load_a16(hbuf, 512, (tm0 + 1) * 16, k0, lane);
#pragma unroll
                for (int j = 0; j < 4; ++j) {
                    v16h b = load_b16(whh16, 512, (tn0 + j) * 16, k0, lane);
                    acc[j]     = wmma16(a0, b, acc[j]);
                    acc[4 + j] = wmma16(a1, b, acc[4 + j]);
                }
            }
#pragma unroll
            for (int mi = 0; mi < 2; ++mi) {
                int mbase = (tm0 + mi) * 16 + ((lane >> 4) & 1) * 8;
#pragma unroll
                for (int j = 0; j < 4; ++j) {
                    int n = (tn0 + j) * 16 + (lane & 15);
                    float bias = bih[n] + bhh[n];
#pragma unroll
                    for (int r = 0; r < 8; ++r)
                        gates[(size_t)(mbase + r) * 2048 + n] = acc[mi * 4 + j][r] + bias;
                }
            }
        }
        grid_barrier(bar);

        // ---- Stage D: LSTM elementwise
        for (int idx = tid; idx < BS * 512; idx += NBLK * NTHR) {
            int b = idx >> 9, j = idx & 511;
            const float* g = gates + (size_t)b * 2048;
            float ig = sigmoidf_(g[j]);
            float fg = sigmoidf_(g[512 + j]);
            float gg = tanhf(g[1024 + j]);
            float og = sigmoidf_(g[1536 + j]);
            float cc = fg * cbuf[idx] + ig * gg;
            cbuf[idx] = cc;
            hbuf[idx] = (_Float16)(og * tanhf(cc));
        }
        grid_barrier(bar);

        // ---- Stage E: blend2 = h @ W2_w   (one 16x16 tile per wave)
        {
            int tm = wave >> 4, tn = wave & 15;
            v8f acc = vzero;
            for (int k0 = 0; k0 < 512; k0 += 32) {
                v16h a = load_a16(hbuf, 512, tm * 16, k0, lane);
                v16h b = load_b16(watT, 512, tn * 16, k0, lane);
                acc = wmma16(a, b, acc);
            }
            int n     = tn * 16 + (lane & 15);
            int mbase = tm * 16 + ((lane >> 4) & 1) * 8;
#pragma unroll
            for (int r = 0; r < 8; ++r)
                bl2[(size_t)(mbase + r) * 256 + n] = acc[r];
        }
        grid_barrier(bar);

        // ---- Stage F: scores, masked log-softmax, argmax, outputs
        {
            const int b = wave;
            const float* bb = bl2 + (size_t)b * 256;
            const int u0 = lane, u1 = lane + 32;
            float s0 = 0.f, s1 = 0.f;
            for (int k = 0; k < 256; ++k) {
                float x = bb[k], w = vt[k];
                s0 += tanhf(b1[u0 * 256 + k] + x) * w;
                s1 += tanhf(b1[u1 * 256 + k] + x) * w;
            }
            float m0 = mask[b * VL + u0] ? NEGV : s0;
            float m1 = mask[b * VL + u1] ? NEGV : s1;

            float bv = (m0 >= m1) ? m0 : m1;     // argmax == softmax max
            int   bi = (m0 >= m1) ? u0 : u1;
#pragma unroll
            for (int off = 16; off > 0; off >>= 1) {
                float ov = __shfl_xor(bv, off);
                int   oi = __shfl_xor(bi, off);
                if (ov > bv || (ov == bv && oi < bi)) { bv = ov; bi = oi; }
            }
            float e = __expf(m0 - bv) + __expf(m1 - bv);
#pragma unroll
            for (int off = 16; off > 0; off >>= 1) e += __shfl_xor(e, off);
            float lz = __logf(e);

            float* lp = out_logp + ((size_t)t * BS + b) * VL;
            lp[u0] = m0 - bv - lz;
            lp[u1] = m1 - bv - lz;
            if (lane == 0) {
                out_views[t * BS + b] = (float)bi;
                vptr[b] = bi;
            }
        }
        grid_barrier(bar);
    }
}

// ----------------------------- launcher ------------------------------------
extern "C" void kernel_launch(void* const* d_in, const int* in_sizes, int n_in,
                              void* d_out, int out_size, void* d_ws, size_t ws_size,
                              hipStream_t stream) {
    if (ws_size < WS_END) return;

    const float* bd   = (const float*)d_in[0];
    const float* enc  = (const float*)d_in[1];
    const float* W0w  = (const float*)d_in[2];
    const float* W0b  = (const float*)d_in[3];
    const float* w1w  = (const float*)d_in[4];
    const float* w1b  = (const float*)d_in[5];
    const float* w2w  = (const float*)d_in[6];
    const float* w2b  = (const float*)d_in[7];
    const float* wih  = (const float*)d_in[8];
    const float* whh  = (const float*)d_in[9];
    const float* bih  = (const float*)d_in[10];
    const float* bhh  = (const float*)d_in[11];
    const float* W1w  = (const float*)d_in[12];
    const float* W2w  = (const float*)d_in[13];
    const float* vt   = (const float*)d_in[14];
    char*  ws  = (char*)d_ws;
    float* out = (float*)d_out;

    // zero barrier/v/mask region and h/c state (fresh every call -> deterministic)
    (void)hipMemsetAsync(ws, 0, O_B1, stream);
    (void)hipMemsetAsync(ws + O_H, 0, O_G - O_H, stream);

    rnnctl_setup<<<256, 256, 0, stream>>>(enc, w1w, w2w, wih, whh, W1w, W2w, ws);

    rnnctl_persistent<<<NBLK, NTHR, 0, stream>>>(bd, W0w, W0b, w1w, w1b, w2b,
                                                 bih, bhh, vt, ws, out);
}
